// EdgeClassification_5317169512700
// MI455X (gfx1250) — matrix operand order repaired and checked by reading, not compiled, not run on hardware
//
#include <hip/hip_runtime.h>
#include <hip/hip_bf16.h>
#include <stdint.h>

#define N_NODES 100000
#define N_EDGES 500000
#define HID 128

typedef __bf16 bf16_t;
typedef __attribute__((ext_vector_type(16))) __bf16 v16bf;
typedef __attribute__((ext_vector_type(8)))  __bf16 v8bf;
typedef __attribute__((ext_vector_type(8)))  float  v8f;
typedef __attribute__((ext_vector_type(4)))  float  v4f;

// f32 -> bf16: native lowering (v_cvt_pk_bf16_f32 confirmed on gfx1250)
__device__ __forceinline__ bf16_t f2bf(float f) { return (bf16_t)f; }

// ---------------- prep kernels ----------------

__global__ void init_minmax_kernel(unsigned* mm) {
  mm[0] = 0xFFFFFFFFu;  // min
  mm[1] = 0u;           // max
}

__global__ __launch_bounds__(256)
void edge_minmax_kernel(const long long* __restrict__ ei, unsigned* __restrict__ mm) {
  __shared__ unsigned smin[256], smax[256];
  unsigned lmin = 0xFFFFFFFFu, lmax = 0u;
  int tid = blockIdx.x * blockDim.x + threadIdx.x;
  int stride = gridDim.x * blockDim.x;
  for (int e = tid; e < N_EDGES; e += stride) {
    long long s = ei[2*e], d = ei[2*e+1];
    long long ad = s > d ? s - d : d - s;
    unsigned a = (unsigned)ad;
    lmin = a < lmin ? a : lmin;
    lmax = a > lmax ? a : lmax;
  }
  smin[threadIdx.x] = lmin; smax[threadIdx.x] = lmax;
  __syncthreads();
  for (int o = 128; o > 0; o >>= 1) {
    if ((int)threadIdx.x < o) {
      unsigned a = smin[threadIdx.x + o]; if (a < smin[threadIdx.x]) smin[threadIdx.x] = a;
      unsigned b = smax[threadIdx.x + o]; if (b > smax[threadIdx.x]) smax[threadIdx.x] = b;
    }
    __syncthreads();
  }
  if (threadIdx.x == 0) {
    atomicMin(&mm[0], smin[0]);
    atomicMax(&mm[1], smax[0]);
  }
}

// W1 [640][256] f32 -> w1t bf16 [N=256][K=512] (transposed, K truncated to 512)
// s1[n] = sum_{k=512..639} W1[k][n]   (folds the constant distance block)
// W2 [256][128] f32 -> w2t bf16 [N=128][K=256]
__global__ __launch_bounds__(256)
void prep_weights_kernel(const float* __restrict__ W1, const float* __restrict__ W2,
                         bf16_t* __restrict__ w1t, bf16_t* __restrict__ w2t,
                         float* __restrict__ s1f) {
  int tid = blockIdx.x * blockDim.x + threadIdx.x;
  int nt  = gridDim.x * blockDim.x;
  for (int i = tid; i < 256 * 512; i += nt) {
    int n = i >> 9, k = i & 511;
    w1t[n * 512 + k] = f2bf(W1[(size_t)k * 256 + n]);
  }
  for (int i = tid; i < 128 * 256; i += nt) {
    int n = i >> 8, k = i & 255;
    w2t[n * 256 + k] = f2bf(W2[(size_t)k * 128 + n]);
  }
  for (int n = tid; n < 256; n += nt) {
    float s = 0.f;
    for (int k = 512; k < 640; k++) s += W1[(size_t)k * 256 + n];
    s1f[n] = s;
  }
}

// ---------------- fragment loaders ----------------

// 16-bit A 16x32 layout: lane L holds row M=L%16; half=L/16 selects
// K in {8h..8h+7} (elems 0..7) and {16+8h..16+8h+7} (elems 8..15).
__device__ __forceinline__ v16bf load_a_lds(const bf16_t* rowp, int half) {
  v8bf lo = *(const v8bf*)(rowp + half * 8);
  v8bf hi = *(const v8bf*)(rowp + 16 + half * 8);
  return __builtin_shufflevector(lo, hi, 0,1,2,3,4,5,6,7,8,9,10,11,12,13,14,15);
}

// Same layout, gathered straight from an f32 node-table row (L2-resident),
// converted with packed v_cvt_pk_bf16_f32 via vector fptrunc.
__device__ __forceinline__ v16bf load_a_f32(const float* p, int half) {
  v4f f0 = *(const v4f*)(p + half * 8);
  v4f f1 = *(const v4f*)(p + half * 8 + 4);
  v4f f2 = *(const v4f*)(p + 16 + half * 8);
  v4f f3 = *(const v4f*)(p + 16 + half * 8 + 4);
  v8f lo32 = __builtin_shufflevector(f0, f1, 0,1,2,3,4,5,6,7);
  v8f hi32 = __builtin_shufflevector(f2, f3, 0,1,2,3,4,5,6,7);
  v8bf lo = __builtin_convertvector(lo32, v8bf);
  v8bf hi = __builtin_convertvector(hi32, v8bf);
  return __builtin_shufflevector(lo, hi, 0,1,2,3,4,5,6,7,8,9,10,11,12,13,14,15);
}

// ---------------- main fused MLP kernel ----------------
// Block = 64 edges, 256 threads = 8 waves (wave32).
// Wave grid: 2 m-groups x 4 n-groups.
//   wave w: mg = w&1 -> m-tiles {2mg, 2mg+1};  ng = w>>1 -> n cols [ng*64, ng*64+64)
// Each B fragment is loaded once per wave and reused for TWO WMMAs.

#define H1_STRIDE 264   // 256 + 8 bf16 pad (528B row: 16B aligned, 4-bank skew)
#define H2_STRIDE 136   // 128 + 8 bf16 pad

__global__ __launch_bounds__(256)
void edge_mlp_kernel(const float* __restrict__ node_emb,
                     const float* __restrict__ type_emb,
                     const long long* __restrict__ ei,
                     const float* __restrict__ b1,
                     const float* __restrict__ b2,
                     const float* __restrict__ W3,
                     const float* __restrict__ b3,
                     const bf16_t* __restrict__ w1t,
                     const bf16_t* __restrict__ w2t,
                     const float* __restrict__ s1f,
                     const unsigned* __restrict__ mm,
                     float* __restrict__ out) {
  __shared__ bf16_t    h1_lds[64 * H1_STRIDE];
  __shared__ bf16_t    h2_lds[64 * H2_STRIDE];
  __shared__ long long ei_lds[128];   // (src,dst) pairs for this block's 64 edges
  __shared__ float     d_lds[64];

  const int tid  = threadIdx.x;
  const int wave = tid >> 5;
  const int lane = tid & 31;
  const int half = lane >> 4;
  const int nl   = lane & 15;
  const int mg   = wave & 1;
  const int ng   = wave >> 1;
  const long long e0 = (long long)blockIdx.x * 64;

  const float dminf  = (float)mm[0];
  const float dscale = 1.0f / ((float)mm[1] - dminf);

  // stage 1: async-copy edge pairs straight into LDS (GLOBAL_LOAD_ASYNC_TO_LDS,
  // ASYNCcnt path) -- one b128 per lane, no VGPR round trip.
  if (tid < 64) {
    long long eg = e0 + tid;
    if (eg >= N_EDGES) eg = N_EDGES - 1;  // tail: compute garbage, don't store
    // generic LDS pointer low 32 bits == workgroup LDS byte offset (ISA 10.2)
    unsigned lds_off = (unsigned)(uintptr_t)(&ei_lds[0]) + (unsigned)tid * 16u;
    unsigned voff    = (unsigned)(eg << 4);  // byte offset of edge pair
    asm volatile("global_load_async_to_lds_b128 %0, %1, %2"
                 :: "v"(lds_off), "v"(voff), "s"(ei)
                 : "memory");
  }
  asm volatile("s_wait_asynccnt 0x0" ::: "memory");
  __syncthreads();

  // normalized |src-dst| per edge
  if (tid < 64) {
    long long s = ei_lds[2 * tid], d = ei_lds[2 * tid + 1];
    long long ad = s > d ? s - d : d - s;
    d_lds[tid] = ((float)ad - dminf) * dscale;
  }
  __syncthreads();

  // this lane's two edge rows (one per owned m-tile)
  const int rowA = mg * 32 + nl;
  const int rowB = rowA + 16;
  const size_t sA = (size_t)(int)ei_lds[2 * rowA] * HID;
  const size_t dA = (size_t)(int)ei_lds[2 * rowA + 1] * HID;
  const size_t sB = (size_t)(int)ei_lds[2 * rowB] * HID;
  const size_t dB = (size_t)(int)ei_lds[2 * rowB + 1] * HID;

  // ---- layer 1: [64 x 512]bf16 @ [512 x 256]bf16 -> f32, +b1 + d*s1, relu
  {
    v8f acc[2][4];
    #pragma unroll
    for (int g = 0; g < 2; g++)
      #pragma unroll
      for (int j = 0; j < 4; j++) acc[g][j] = (v8f)(0.0f);

    // fully unrolled over the 4 fused segments so each segment's base pointer
    // is a compile-time selection -> pure global_load_b128 (no FLAT, no
    // pointer array, no joint loadcnt/dscnt waits in the hot loop)
    #pragma unroll
    for (int seg = 0; seg < 4; seg++) {
      const float* pA = (seg == 0) ? node_emb + sA
                      : (seg == 1) ? node_emb + dA
                      : (seg == 2) ? type_emb + sA
                                   : type_emb + dA;
      const float* pB = (seg == 0) ? node_emb + sB
                      : (seg == 1) ? node_emb + dB
                      : (seg == 2) ? type_emb + sB
                                   : type_emb + dB;
      #pragma unroll
      for (int ko = 0; ko < 128; ko += 32) {
        const int kc = seg * 128 + ko;
        v16bf a0 = load_a_f32(pA + ko, half);
        v16bf a1 = load_a_f32(pB + ko, half);
        #pragma unroll
        for (int j = 0; j < 4; j++) {
          int n = ng * 64 + j * 16 + nl;
          const bf16_t* bp = w1t + (size_t)n * 512 + kc + half * 8;
          v16bf b = __builtin_shufflevector(*(const v8bf*)bp, *(const v8bf*)(bp + 16), 0,1,2,3,4,5,6,7,8,9,10,11,12,13,14,15);
          acc[0][j] = __builtin_amdgcn_wmma_f32_16x16x32_bf16(
              false, a0, false, b, (short)0, acc[0][j], false, false);
          acc[1][j] = __builtin_amdgcn_wmma_f32_16x16x32_bf16(
              false, a1, false, b, (short)0, acc[1][j], false, false);
        }
      }
    }
    #pragma unroll
    for (int g = 0; g < 2; g++) {
      #pragma unroll
      for (int j = 0; j < 4; j++) {
        int n = ng * 64 + j * 16 + nl;
        float bias = b1[n];
        float sv   = s1f[n];
        #pragma unroll
        for (int r = 0; r < 8; r++) {
          int m = mg * 32 + g * 16 + r + 8 * half;  // D layout: M = r + 8*(lane/16)
          float v = acc[g][j][r] + bias + d_lds[m] * sv;
          h1_lds[m * H1_STRIDE + n] = f2bf(v > 0.0f ? v : 0.0f);
        }
      }
    }
  }
  __syncthreads();

  // ---- layer 2: [64 x 256]bf16 @ [256 x 128]bf16 -> f32, +b2, relu
  {
    v8f acc[2][2];
    #pragma unroll
    for (int g = 0; g < 2; g++)
      #pragma unroll
      for (int j = 0; j < 2; j++) acc[g][j] = (v8f)(0.0f);

    const bf16_t* arow0 = h1_lds + rowA * H1_STRIDE;
    const bf16_t* arow1 = h1_lds + rowB * H1_STRIDE;
    for (int kc = 0; kc < 256; kc += 32) {
      v16bf a0 = load_a_lds(arow0 + kc, half);
      v16bf a1 = load_a_lds(arow1 + kc, half);
      #pragma unroll
      for (int j = 0; j < 2; j++) {
        int n = ng * 32 + j * 16 + nl;
        const bf16_t* bp = w2t + (size_t)n * 256 + kc + half * 8;
        v16bf b = __builtin_shufflevector(*(const v8bf*)bp, *(const v8bf*)(bp + 16), 0,1,2,3,4,5,6,7,8,9,10,11,12,13,14,15);
        acc[0][j] = __builtin_amdgcn_wmma_f32_16x16x32_bf16(
            false, a0, false, b, (short)0, acc[0][j], false, false);
        acc[1][j] = __builtin_amdgcn_wmma_f32_16x16x32_bf16(
            false, a1, false, b, (short)0, acc[1][j], false, false);
      }
    }
    #pragma unroll
    for (int g = 0; g < 2; g++) {
      #pragma unroll
      for (int j = 0; j < 2; j++) {
        int n = ng * 32 + j * 16 + nl;
        float bias = b2[n];
        #pragma unroll
        for (int r = 0; r < 8; r++) {
          int m = mg * 32 + g * 16 + r + 8 * half;
          float v = acc[g][j][r] + bias;
          h2_lds[m * H2_STRIDE + n] = f2bf(v > 0.0f ? v : 0.0f);
        }
      }
    }
  }
  __syncthreads();

  // ---- layer 3: [64 x 128] @ [128 x 5] + b3 (tiny; vectorized VALU)
  for (int idx = tid; idx < 64 * 5; idx += 256) {
    int e = idx / 5, c = idx - e * 5;
    long long eg = e0 + e;
    if (eg < N_EDGES) {
      float accv = b3[c];
      const bf16_t* hp = h2_lds + e * H2_STRIDE;
      #pragma unroll
      for (int k = 0; k < 128; k += 8) {
        v8bf hb = *(const v8bf*)(hp + k);
        v8f  hf = __builtin_convertvector(hb, v8f);
        #pragma unroll
        for (int u = 0; u < 8; u++) accv += hf[u] * W3[(k + u) * 5 + c];
      }
      out[eg * 5 + c] = accv;
    }
  }
}

// ---------------- launch ----------------

extern "C" void kernel_launch(void* const* d_in, const int* in_sizes, int n_in,
                              void* d_out, int out_size, void* d_ws, size_t ws_size,
                              hipStream_t stream) {
  const float*     node_emb = (const float*)d_in[0];
  const float*     type_emb = (const float*)d_in[1];
  const long long* ei       = (const long long*)d_in[2];
  const float*     W1       = (const float*)d_in[3];
  const float*     b1       = (const float*)d_in[4];
  const float*     W2       = (const float*)d_in[5];
  const float*     b2       = (const float*)d_in[6];
  const float*     W3       = (const float*)d_in[7];
  const float*     b3       = (const float*)d_in[8];
  float* out = (float*)d_out;

  // workspace layout (all offsets 32B-aligned for v16bf loads)
  unsigned* mm  = (unsigned*)d_ws;
  float*    s1f = (float*)((char*)d_ws + 64);
  bf16_t*   w1t = (bf16_t*)((char*)d_ws + 64 + 1024);
  bf16_t*   w2t = (bf16_t*)((char*)d_ws + 64 + 1024 + 256 * 512 * 2);

  init_minmax_kernel<<<1, 1, 0, stream>>>(mm);
  edge_minmax_kernel<<<256, 256, 0, stream>>>(ei, mm);
  prep_weights_kernel<<<256, 256, 0, stream>>>(W1, W2, w1t, w2t, s1f);

  int nblocks = (N_EDGES + 63) / 64;
  edge_mlp_kernel<<<nblocks, 256, 0, stream>>>(
      node_emb, type_emb, ei, b1, b2, W3, b3, w1t, w2t, s1f, mm, out);
}
